// VanillaRNN_7164005449728
// MI455X (gfx1250) — matrix operand order repaired and checked by reading, not compiled, not run on hardware
//
#include <hip/hip_runtime.h>

// Problem constants (match reference)
constexpr int kB = 256;   // batch
constexpr int kT = 128;   // time steps
constexpr int kH = 2048;  // hidden
constexpr int kC = 10;    // classes

// TDM / tiling constants
constexpr int kChunkK  = 32;                 // K elements staged per TDM chunk
constexpr int kRowPadH = 8;                  // 8 f16 = 16B pad per LDS row
constexpr int kLdsRowH = kChunkK + kRowPadH; // 40 f16 = 80B LDS row stride
constexpr int kBlkN    = 256;                // block N extent
constexpr int kBlkM    = 64;                 // block M extent

typedef __attribute__((ext_vector_type(16))) _Float16 v16h;
typedef __attribute__((ext_vector_type(8)))  _Float16 v8h;
typedef __attribute__((ext_vector_type(8)))  float    v8f;
typedef __attribute__((ext_vector_type(4)))  unsigned v4u;
typedef __attribute__((ext_vector_type(8)))  int      v8i;
typedef __attribute__((ext_vector_type(4)))  int      v4i;

// ---------------------------------------------------------------------------
// Issue one TDM DMA: a (kChunkK x kBlkN) f16 tile of w_hh (row-major, row
// stride kH) from global memory into LDS, with 16B padding per 64B row so
// subsequent ds_load_b128 reads are bank-conflict free. D# layout per
// CDNA5 ISA §8 (group0: count/lds/global/type, group1: dims/strides/pad).
// ---------------------------------------------------------------------------
__device__ __forceinline__ void tdm_load_b_tile(const _Float16* gsrc,
                                                const _Float16* lds_dst) {
  unsigned long long ga = (unsigned long long)(uintptr_t)gsrc;
  unsigned lds_off = (unsigned)(uintptr_t)lds_dst;   // low 32 bits = LDS offset

  v4u g0;
  g0.x = 1u;                                    // count=1, user descriptor
  g0.y = lds_off;                               // lds_addr (bytes)
  g0.z = (unsigned)(ga & 0xffffffffu);          // global_addr[31:0]
  g0.w = (unsigned)((ga >> 32) & 0x01ffffffu)   // global_addr[56:32]
       | (2u << 30);                            // type = 2 ("image")

  unsigned long long q0 =
        (1ull << 16)                            // data_size: 1 -> 2 bytes
      | (1ull << 20)                            // pad_enable
      | (3ull << 22)                            // pad_interval: 16 DWORDs (64B)
      | (3ull << 25)                            // pad_amount: 4 DWORDs (16B)
      | ((unsigned long long)(kH & 0xffff) << 48);      // tensor_dim0[15:0]
  unsigned long long q1 =
        ((unsigned long long)kH >> 16)          // tensor_dim0[31:16]
      | ((unsigned long long)kH << 16)          // tensor_dim1
      | ((unsigned long long)kChunkK << 48);    // tile_dim0
  unsigned long long q2 =
        (unsigned long long)kBlkN               // tile_dim1 (tile_dim2 = 0)
      | (((unsigned long long)kH & 0xffffffffull) << 32); // dim0_stride[31:0]
  unsigned long long q3 =
        ((unsigned long long)kH >> 32);         // dim0_stride[47:32]=0, dim1_stride=0

  v8i g1;
  g1[0] = (int)(unsigned)q0; g1[1] = (int)(unsigned)(q0 >> 32);
  g1[2] = (int)(unsigned)q1; g1[3] = (int)(unsigned)(q1 >> 32);
  g1[4] = (int)(unsigned)q2; g1[5] = (int)(unsigned)(q2 >> 32);
  g1[6] = (int)(unsigned)q3; g1[7] = (int)(unsigned)(q3 >> 32);

  v4i gz = {0, 0, 0, 0};
#if defined(__clang_major__) && (__clang_major__ >= 23)
  v8i gz8 = {0, 0, 0, 0, 0, 0, 0, 0};
  __builtin_amdgcn_tensor_load_to_lds(g0, g1, gz, gz, gz8, 0);
#else
  __builtin_amdgcn_tensor_load_to_lds(g0, g1, gz, gz, 0);
#endif
}

// ---------------------------------------------------------------------------
// Cast w_hh (f32, HxH) to f16 once; 8 MB lives in L2 for all 128 steps.
// ---------------------------------------------------------------------------
__global__ __launch_bounds__(256) void rnn_cast_w(const float* __restrict__ w,
                                                  _Float16* __restrict__ w16) {
    size_t i = (size_t)blockIdx.x * 256 + threadIdx.x;
    w16[i] = (_Float16)w[i];
}

__global__ __launch_bounds__(256) void rnn_zero_h(_Float16* __restrict__ h) {
    size_t i = (size_t)blockIdx.x * 256 + threadIdx.x;
    h[i] = (_Float16)0.0f;
}

// ---------------------------------------------------------------------------
// One recurrence step: hnext = tanh(x[:,t]*wx + hprev @ w_hh^T + b_h)
// GEMM M=256, N=2048, K=2048.
// Block: 256 threads = 8 waves (2M x 4N); wave tile = 32M x 64N (8 accums).
// Block tile 64M x 256N; grid (4, 8). B staged in LDS by TDM, double-buffered.
// ---------------------------------------------------------------------------
__global__ __launch_bounds__(256) void rnn_step(
    const _Float16* __restrict__ hprev,  // [B][H] f16
    const _Float16* __restrict__ w16,    // [H][H] f16 (w_hh row-major)
    const float*    __restrict__ x,      // [B][T]
    const float*    __restrict__ wx,     // [H]   (w_hx[:,0])
    const float*    __restrict__ bh,     // [H]
    _Float16*       __restrict__ hnext,  // [B][H] f16
    int t)
{
    __shared__ __align__(16) _Float16 bbuf[2][kBlkN * kLdsRowH]; // 2 x 20 KB

    const int lane = threadIdx.x & 31;
    const int wave = threadIdx.x >> 5;          // 0..7
    const int lm   = lane & 15;
    const int hi   = lane >> 4;                 // half-wave select

    const int m0    = blockIdx.x * kBlkM + (wave >> 2) * 32;  // batch base
    const int n0blk = blockIdx.y * kBlkN;                     // block hidden base
    const int nw    = (wave & 3) * 64;                        // wave N offset in block

    const v8f vz = {0.f, 0.f, 0.f, 0.f, 0.f, 0.f, 0.f, 0.f};
    v8f acc[2][4];
    #pragma unroll
    for (int mt = 0; mt < 2; ++mt)
      #pragma unroll
      for (int j = 0; j < 4; ++j) acc[mt][j] = vz;

    // Global source of the first B tile: element (k=0, n=n0blk)
    const _Float16* bsrc = w16 + (size_t)n0blk * kH;

    if (wave == 0) tdm_load_b_tile(bsrc, &bbuf[0][0]);

    const int nchunks = kH / kChunkK;   // 64
    for (int c = 0; c < nchunks; ++c) {
        if (wave == 0) {
            if (c + 1 < nchunks) {
                tdm_load_b_tile(bsrc + (size_t)(c + 1) * kChunkK,
                                &bbuf[(c + 1) & 1][0]);
                __builtin_amdgcn_s_wait_tensorcnt(1);  // chunk c complete
            } else {
                __builtin_amdgcn_s_wait_tensorcnt(0);
            }
        }
        __syncthreads();                 // B tile for chunk c visible to all

        const int kg = c * kChunkK;      // global K base of this chunk
        const _Float16* bbase = &bbuf[c & 1][0];

        // A-matrix 16x32 f16 layout (ISA 7.12.2): lane half selects K groups.
        v16h a[2];
        #pragma unroll
        for (int mt = 0; mt < 2; ++mt) {
            const _Float16* ar =
                hprev + (size_t)(m0 + mt * 16 + lm) * kH + kg + hi * 16;
            v8h alo = *(const v8h*)ar;
            v8h ahi = *(const v8h*)(ar + 8);
            #pragma unroll
            for (int i = 0; i < 8; ++i) { a[mt][i] = alo[i]; a[mt][i + 8] = ahi[i]; }
        }

        // B-matrix 32x16 f16 layout: lane holds 16 contiguous K (hi*16 base),
        // lane%16 selects column n. Read from padded LDS rows.
        #pragma unroll
        for (int j = 0; j < 4; ++j) {
            const _Float16* bp =
                bbase + (size_t)(nw + j * 16 + lm) * kLdsRowH + hi * 16;
            v8h blo = *(const v8h*)bp;
            v8h bhi = *(const v8h*)(bp + 8);
            v16h b;
            #pragma unroll
            for (int i = 0; i < 8; ++i) { b[i] = blo[i]; b[i + 8] = bhi[i]; }

            acc[0][j] = __builtin_amdgcn_wmma_f32_16x16x32_f16(
                            false, a[0], false, b, (short)0, acc[0][j], false, false);
            acc[1][j] = __builtin_amdgcn_wmma_f32_16x16x32_f16(
                            false, a[1], false, b, (short)0, acc[1][j], false, false);
        }
        __syncthreads();                 // all reads of bbuf[c&1] done
    }

    // Epilogue: C/D 16x16 f32 layout — VGPR r = row (r + 8*hi), lane%16 = col.
    #pragma unroll
    for (int mt = 0; mt < 2; ++mt) {
        #pragma unroll
        for (int j = 0; j < 4; ++j) {
            const int n = n0blk + nw + j * 16 + lm;
            const float wxv = wx[n];
            const float bhv = bh[n];
            #pragma unroll
            for (int r = 0; r < 8; ++r) {
                const int bi = m0 + mt * 16 + hi * 8 + r;
                float pre = acc[mt][j][r] + x[(size_t)bi * kT + t] * wxv + bhv;
                hnext[(size_t)bi * kH + n] = (_Float16)tanhf(pre);
            }
        }
    }
}

// ---------------------------------------------------------------------------
// Final projection: out[b][c] = sum_k h[b][k] * w_ph[c][k] + b_p[c]
// ---------------------------------------------------------------------------
__global__ __launch_bounds__(256) void rnn_proj(
    const _Float16* __restrict__ h,    // [B][H] f16
    const float*    __restrict__ wph,  // [C][H]
    const float*    __restrict__ bp,   // [C]
    float*          __restrict__ out)  // [B][C]
{
    const int b   = blockIdx.x;
    const int tid = threadIdx.x;
    __shared__ float red[kC][256];

    float p[kC];
    #pragma unroll
    for (int c = 0; c < kC; ++c) p[c] = 0.0f;

    for (int k = tid; k < kH; k += 256) {
        float hv = (float)h[(size_t)b * kH + k];
        #pragma unroll
        for (int c = 0; c < kC; ++c) p[c] += hv * wph[(size_t)c * kH + k];
    }
    #pragma unroll
    for (int c = 0; c < kC; ++c) red[c][tid] = p[c];
    __syncthreads();

    for (int s = 128; s > 0; s >>= 1) {
        if (tid < s) {
            #pragma unroll
            for (int c = 0; c < kC; ++c) red[c][tid] += red[c][tid + s];
        }
        __syncthreads();
    }
    if (tid < kC) out[(size_t)b * kC + tid] = red[tid][0] + bp[tid];
}

// ---------------------------------------------------------------------------
// Host-side orchestration (graph-capture safe: only kernel launches on stream)
// ---------------------------------------------------------------------------
extern "C" void kernel_launch(void* const* d_in, const int* in_sizes, int n_in,
                              void* d_out, int out_size, void* d_ws, size_t ws_size,
                              hipStream_t stream) {
    (void)in_sizes; (void)n_in; (void)out_size; (void)ws_size;

    const float* x    = (const float*)d_in[0];  // [B][T]
    const float* w_hx = (const float*)d_in[1];  // [H][1]
    const float* w_hh = (const float*)d_in[2];  // [H][H]
    const float* w_ph = (const float*)d_in[3];  // [C][H]
    const float* b_h  = (const float*)d_in[4];  // [H]
    const float* b_p  = (const float*)d_in[5];  // [C]
    float* out = (float*)d_out;

    // Workspace layout: w16 (8 MB) | hA (1 MB) | hB (1 MB)
    char* ws = (char*)d_ws;
    _Float16* w16 = (_Float16*)ws;
    _Float16* hA  = (_Float16*)(ws + (size_t)kH * kH * sizeof(_Float16));
    _Float16* hB  = hA + (size_t)kB * kH;

    rnn_cast_w<<<dim3((kH * kH) / 256), dim3(256), 0, stream>>>(w_hh, w16);
    rnn_zero_h<<<dim3((kB * kH) / 256), dim3(256), 0, stream>>>(hA);

    _Float16* cur = hA;
    _Float16* nxt = hB;
    for (int t = 0; t < kT; ++t) {
        rnn_step<<<dim3(kB / kBlkM, kH / kBlkN), dim3(256), 0, stream>>>(
            cur, w16, x, w_hx, b_h, nxt, t);
        _Float16* tmp = cur; cur = nxt; nxt = tmp;
    }

    rnn_proj<<<dim3(kB), dim3(256), 0, stream>>>(cur, w_ph, b_p, out);
}
